// DefectPredictorModel_63316407878109
// MI455X (gfx1250) — matrix operand order repaired and checked by reading, not compiled, not run on hardware
//
#include <hip/hip_runtime.h>
#include <hip/hip_bf16.h>
#include <cstdint>

// DefectPredictor GNN for gfx1250 (MI455X). Dense node GEMMs use
// v_wmma_f32_16x16x32_bf16; each wave computes a 16x32 output strip
// (shared A fragment, two accumulators, branch-free clamped loads).
// Linear edge-attention term folded to a 40x4 matrix; geo-MLP fused with
// weights in LDS. Workspace use: ~310 MB.

#define H   128
#define NH  4
#define HD  32
#define BINS 40
#define NL  3

typedef __bf16 bf16_t;
typedef __attribute__((ext_vector_type(16))) bf16_t v16bf;
typedef __attribute__((ext_vector_type(8)))  bf16_t v8bf;
typedef __attribute__((ext_vector_type(8)))  float  v8f;

__device__ __forceinline__ unsigned short f2bf(float f) {
  unsigned u = __float_as_uint(f);
  u += 0x7FFFu + ((u >> 16) & 1u);          // round-to-nearest-even
  return (unsigned short)(u >> 16);
}
__device__ __forceinline__ float bf2f(unsigned short b) {
  return __uint_as_float(((unsigned)b) << 16);
}
__device__ __forceinline__ float silu_f(float x) { return x / (1.f + __expf(-x)); }
__device__ __forceinline__ void atomicMaxF(float* addr, float val) {
  int* ai = (int*)addr;
  int cur = *ai;
  while (__int_as_float(cur) < val) {
    int old = atomicCAS(ai, cur, __float_as_int(val));
    if (old == cur) break;
    cur = old;
  }
}
#define GIDX() ((long long)blockIdx.x * blockDim.x + threadIdx.x)

// ---------------- utility kernels ----------------
__global__ void k_fill(float* p, float v, long long n) {
  long long i = GIDX(); if (i < n) p[i] = v;
}
__global__ void k_tobf16(const float* s, unsigned short* d, long long n) {
  long long i = GIDX(); if (i < n) d[i] = f2bf(s[i]);
}
__global__ void k_transpose_bf16(const float* W, unsigned short* Wt, int K, int Nn) {
  int i = blockIdx.x * blockDim.x + threadIdx.x;
  if (i >= K * Nn) return;
  int k = i / Nn, n = i % Nn;
  Wt[(size_t)n * K + k] = f2bf(W[i]);
}

// ---------------- WMMA bf16 GEMM ----------------
// C[M,Nn] = A[M,K](bf16, row-major) x B  (given as Bt=[Nn,K] bf16) + bias.
// One wave per 16x32 output strip: shared A fragment, two accumulators.
// A rows are clamped (not masked) for the M tail: garbage rows only affect
// D rows that are never stored (rows of D are independent in WMMA).
__global__ void __launch_bounds__(32)
k_gemm_bf16(const unsigned short* A, const unsigned short* Bt, float* C,
            const float* bias, int M, int K, int Nn) {
  int tm = blockIdx.x, tn = blockIdx.y;          // tn covers 32 columns
  int lane = threadIdx.x;
  int half = lane >> 4, l16 = lane & 15;
  int rowA = tm * 16 + l16;
  int ra = rowA < M ? rowA : (M - 1);            // clamped, branch-free loads
  const bf16_t* ap  = (const bf16_t*)(A + (size_t)ra * K);
  const bf16_t* bp0 = (const bf16_t*)(Bt + (size_t)(tn * 32 + l16) * K);
  const bf16_t* bp1 = (const bf16_t*)(Bt + (size_t)(tn * 32 + 16 + l16) * K);
  v8f acc0 = {}, acc1 = {};
  for (int k0 = 0; k0 < K; k0 += 32) {
    // A 16x32 bf16, contiguous k order used here (B matches): element e -> K = e + 16*half, M = lane%16
    v8bf a0 = *(const v8bf*)(ap + k0 + 16 * half);
    v8bf a1 = *(const v8bf*)(ap + k0 + 16 * half + 8);
    v16bf a = __builtin_shufflevector(a0, a1, 0,1,2,3,4,5,6,7,8,9,10,11,12,13,14,15);
    // B 32x16 bf16, same contiguous k order as A: element e -> K = e + 16*half, N = lane%16 (Bt row = contiguous K)
    v8bf b00 = *(const v8bf*)(bp0 + k0 + 16 * half);
    v8bf b01 = *(const v8bf*)(bp0 + k0 + 16 * half + 8);
    v16bf bv0 = __builtin_shufflevector(b00, b01, 0,1,2,3,4,5,6,7,8,9,10,11,12,13,14,15);
    v8bf b10 = *(const v8bf*)(bp1 + k0 + 16 * half);
    v8bf b11 = *(const v8bf*)(bp1 + k0 + 16 * half + 8);
    v16bf bv1 = __builtin_shufflevector(b10, b11, 0,1,2,3,4,5,6,7,8,9,10,11,12,13,14,15);
    acc0 = __builtin_amdgcn_wmma_f32_16x16x32_bf16(false, a, false, bv0, (short)0, acc0,
                                                   false, false);
    acc1 = __builtin_amdgcn_wmma_f32_16x16x32_bf16(false, a, false, bv1, (short)0, acc1,
                                                   false, false);
  }
  int col0 = tn * 32 + l16;
  int col1 = col0 + 16;
  float bv0s = bias ? bias[col0] : 0.f;
  float bv1s = bias ? bias[col1] : 0.f;
  // D layout: VGPR r -> M = r + 8*half, N = lane%16
#pragma unroll
  for (int r = 0; r < 8; r++) {
    int row = tm * 16 + r + 8 * half;
    if (row < M) {
      C[(size_t)row * Nn + col0] = acc0[r] + bv0s;
      C[(size_t)row * Nn + col1] = acc1[r] + bv1s;
    }
  }
}

// ---------------- feature construction ----------------
__global__ void k_rbf(const float* ea, unsigned short* efb, int E) {
  long long i = GIDX();
  if (i >= (long long)E * 64) return;
  int e = (int)(i >> 6), c = (int)(i & 63);
  float v = 0.f;
  if (c < BINS) { float ctr = 8.f * c / (BINS - 1); float d = ea[e] - ctr; v = __expf(-10.f * d * d); }
  efb[i] = f2bf(v);
}
__global__ void k_loop_acc(const float* ea, const int* dst, float* efloop, float* cnt, int E) {
  long long i = GIDX();
  if (i >= (long long)E * BINS) return;
  int e = (int)(i / BINS), b = (int)(i % BINS);
  float ctr = 8.f * b / (BINS - 1); float d = ea[e] - ctr;
  atomicAdd(&efloop[(size_t)dst[e] * BINS + b], __expf(-10.f * d * d));
  if (b == 0) atomicAdd(&cnt[dst[e]], 1.f);
}
__global__ void k_loop_fin(const float* efloop, const float* cnt, unsigned short* efb, int N, int E) {
  long long i = GIDX();
  if (i >= (long long)N * 64) return;
  int n = (int)(i >> 6), c = (int)(i & 63);
  float v = 0.f;
  if (c < BINS) v = efloop[(size_t)n * BINS + c] / fmaxf(cnt[n], 1.f);
  efb[((size_t)(E + n) << 6) + c] = f2bf(v);
}
__global__ void k_embed(const int* x, const int* isdef, const float* aemb, const float* demb,
                        float* h, int N) {
  long long i = GIDX();
  if (i >= (long long)N * H) return;
  int n = (int)(i >> 7), j = (int)(i & 127);
  h[i] = aemb[(size_t)x[n] * H + j] + demb[(size_t)isdef[n] * H + j];
}

// ---------------- GAT attention ----------------
__global__ void k_fold_we(const float* ew, const float* ae, float* we) {
  int i = blockIdx.x * blockDim.x + threadIdx.x;
  if (i >= BINS * NH) return;
  int b = i / NH, hh = i % NH;
  float s = 0.f;
  for (int d = 0; d < HD; d++) s += ew[b * H + hh * HD + d] * ae[hh * HD + d];
  we[b * NH + hh] = s;
}
__global__ void k_node_coef(const float* xl, const float* as_, const float* ad_,
                            float* a_s, float* a_d, int N) {
  long long i = GIDX();
  if (i >= (long long)N * NH) return;
  int n = (int)(i >> 2), hh = (int)(i & 3);
  const float* xr = xl + (size_t)n * H + hh * HD;
  float s = 0.f, d = 0.f;
  for (int t = 0; t < HD; t++) { s += xr[t] * as_[hh * HD + t]; d += xr[t] * ad_[hh * HD + t]; }
  a_s[i] = s; a_d[i] = d;
}
__global__ void k_gat_score(const float* a_s, const float* a_d, const unsigned short* efb,
                            const float* we, const int* src, const int* dst,
                            float* alpha, float* segmax, int E, int n_items) {
  long long i = GIDX();
  if (i >= (long long)n_items * NH) return;
  int it = (int)(i >> 2), hh = (int)(i & 3);
  int s = it < E ? src[it] : it - E;
  int d = it < E ? dst[it] : it - E;
  float v = a_s[s * NH + hh] + a_d[d * NH + hh];
  const unsigned short* er = efb + ((size_t)it << 6);
  for (int b = 0; b < BINS; b++) v += bf2f(er[b]) * we[b * NH + hh];
  v = v > 0.f ? v : 0.2f * v;  // leaky_relu(0.2)
  alpha[i] = v;
  atomicMaxF(&segmax[d * NH + hh], v);
}
__global__ void k_seg_exp(float* alpha, const int* dst, const float* segmax, float* segden,
                          int E, int n_items) {
  long long i = GIDX();
  if (i >= (long long)n_items * NH) return;
  int it = (int)(i >> 2), hh = (int)(i & 3);
  int d = it < E ? dst[it] : it - E;
  float ev = __expf(alpha[i] - segmax[d * NH + hh]);
  alpha[i] = ev;
  atomicAdd(&segden[d * NH + hh], ev);
}
__global__ void k_seg_norm(float* alpha, const int* dst, const float* segden, int E, int n_items) {
  long long i = GIDX();
  if (i >= (long long)n_items * NH) return;
  int it = (int)(i >> 2), hh = (int)(i & 3);
  int d = it < E ? dst[it] : it - E;
  alpha[i] /= (segden[d * NH + hh] + 1e-16f);
}
__global__ void k_scatter(const float* feat, const float* alpha, const int* src, const int* dst,
                          float* out, int E, int n_items) {
  long long i = GIDX();
  if (i >= (long long)n_items * H) return;
  int it = (int)(i >> 7), j = (int)(i & 127);
  int s = it < E ? src[it] : it - E;
  int d = it < E ? dst[it] : it - E;
  float a = alpha[(size_t)it * NH + (j >> 5)];
  atomicAdd(&out[(size_t)d * H + j], feat[(size_t)s * H + j] * a);
}
__global__ void __launch_bounds__(128)
k_update_ln(float* h, const float* add, const float* bias, int do_silu, int N) {
  int n = blockIdx.x, j = threadIdx.x;
  __shared__ float red[H];
  size_t o = (size_t)n * H + j;
  float v = h[o] + add[o] + (bias ? bias[j] : 0.f);
  red[j] = v; __syncthreads();
  for (int s = 64; s > 0; s >>= 1) { if (j < s) red[j] += red[j + s]; __syncthreads(); }
  float mu = red[0] * (1.f / H); __syncthreads();
  float dv = v - mu;
  red[j] = dv * dv; __syncthreads();
  for (int s = 64; s > 0; s >>= 1) { if (j < s) red[j] += red[j + s]; __syncthreads(); }
  float var = red[0] * (1.f / H);
  float out = dv * rsqrtf(var + 1e-5f);
  if (do_silu) out = silu_f(out);
  h[o] = out;
}

// ---------------- transformer attention ----------------
__global__ void __launch_bounds__(128)
k_geo(const unsigned short* efb, const float* gw1, const float* gb1,
      const float* gw2, const float* gb2, float* geo, int E) {
  __shared__ float sw1[BINS * H];
  __shared__ float sw2[H * NH];
  __shared__ float sb1[H];
  __shared__ float sb2[NH];
  __shared__ float sef[BINS];
  __shared__ float sg1[H];
  int j = threadIdx.x;
  for (int t = j; t < BINS * H; t += 128) sw1[t] = gw1[t];
  for (int t = j; t < H * NH; t += 128) sw2[t] = gw2[t];
  sb1[j] = gb1[j];
  if (j < NH) sb2[j] = gb2[j];
  __syncthreads();
  for (int e = blockIdx.x; e < E; e += gridDim.x) {
    if (j < BINS) sef[j] = bf2f(efb[((size_t)e << 6) + j]);
    __syncthreads();
    float a = sb1[j];
    for (int b = 0; b < BINS; b++) a += sef[b] * sw1[b * H + j];
    sg1[j] = silu_f(a);
    __syncthreads();
    if (j < NH) {
      float s = sb2[j];
      for (int t = 0; t < H; t++) s += sg1[t] * sw2[t * NH + j];
      geo[(size_t)e * NH + j] = s;
    }
    __syncthreads();
  }
}
__global__ void k_qk_score(const float* Q, const float* K, const float* geo, const float* dbias,
                           const int* isdef, const int* src, const int* dst,
                           float* alpha, float* segmax, int E) {
  long long i = GIDX();
  if (i >= (long long)E * NH) return;
  int e = (int)(i >> 2), hh = (int)(i & 3);
  int s = src[e], d = dst[e];
  const float* q = Q + (size_t)s * H + hh * HD;
  const float* k = K + (size_t)d * H + hh * HD;
  float sc = 0.f;
  for (int t = 0; t < HD; t++) sc += q[t] * k[t];
  sc *= 0.17677669529663687f;  // 1/sqrt(32)
  int code = isdef[s] * 2 + isdef[d];
  sc += geo[(size_t)e * NH + hh] + dbias[hh * 4 + code];
  alpha[i] = sc;
  atomicMaxF(&segmax[d * NH + hh], sc);
}

// ---------------- pooling + MLP head ----------------
__global__ void k_pool(const float* h, const int* batch, float* pooled, float* gcnt, int N) {
  long long i = GIDX();
  if (i >= (long long)N * H) return;
  int n = (int)(i >> 7), j = (int)(i & 127);
  int g = batch[n];
  atomicAdd(&pooled[(size_t)g * H + j], h[i]);
  if (j == 0) atomicAdd(&gcnt[g], 1.f);
}
__global__ void __launch_bounds__(128)
k_mlp(const float* pooled, const float* gcnt, const float* w1, const float* b1,
      const float* w2, const float* b2, float* out) {
  int g = blockIdx.x, j = threadIdx.x;
  __shared__ float pm[H];
  __shared__ float hid[H];
  pm[j] = pooled[(size_t)g * H + j] / fmaxf(gcnt[g], 1.f);
  __syncthreads();
  float a = b1[j];
  for (int t = 0; t < H; t++) a += pm[t] * w1[t * H + j];
  hid[j] = silu_f(a);
  __syncthreads();
  if (j == 0) {
    float s = b2[0];
    for (int t = 0; t < H; t++) s += hid[t] * w2[t];
    out[g] = s;
  }
}

// ---------------- host orchestration ----------------
extern "C" void kernel_launch(void* const* d_in, const int* in_sizes, int n_in,
                              void* d_out, int out_size, void* d_ws, size_t ws_size,
                              hipStream_t stream) {
  const int*   x          = (const int*)  d_in[0];
  const int*   is_defect  = (const int*)  d_in[1];
  const int*   edge_index = (const int*)  d_in[2];
  const int*   batch      = (const int*)  d_in[3];
  const float* edge_attr  = (const float*)d_in[4];
  const float* atom_emb   = (const float*)d_in[5];
  const float* defect_emb = (const float*)d_in[6];
  const float* gat_w  = (const float*)d_in[7];
  const float* gat_as = (const float*)d_in[8];
  const float* gat_ad = (const float*)d_in[9];
  const float* gat_ew = (const float*)d_in[10];
  const float* gat_ae = (const float*)d_in[11];
  const float* gat_b  = (const float*)d_in[12];
  const float* qw  = (const float*)d_in[13];
  const float* kw  = (const float*)d_in[14];
  const float* vw  = (const float*)d_in[15];
  const float* ow  = (const float*)d_in[16];
  const float* gw1 = (const float*)d_in[17];
  const float* gw2 = (const float*)d_in[18];
  const float* qb  = (const float*)d_in[19];
  const float* kb  = (const float*)d_in[20];
  const float* vb  = (const float*)d_in[21];
  const float* ob  = (const float*)d_in[22];
  const float* gb1 = (const float*)d_in[23];
  const float* gb2 = (const float*)d_in[24];
  const float* dbias = (const float*)d_in[25];
  const float* fcw1  = (const float*)d_in[26];
  const float* fcb1  = (const float*)d_in[27];
  const float* fcw2  = (const float*)d_in[28];
  const float* fcb2  = (const float*)d_in[29];
  (void)n_in; (void)ws_size;

  const int N  = in_sizes[0];
  const int E  = in_sizes[4];
  const int EA = E + N;
  const int G  = out_size;
  const int* src = edge_index;
  const int* dst = edge_index + E;

  char* wsp = (char*)d_ws;
  auto alloc = [&](size_t bytes) -> void* {
    void* p = (void*)wsp;
    wsp += (bytes + 255) & ~(size_t)255;
    return p;
  };
  unsigned short* efb    = (unsigned short*)alloc((size_t)EA * 64 * 2);  // bf16 RBF, padded K=64
  float*          efloop = (float*)alloc((size_t)N * BINS * 4);
  float*          cnt    = (float*)alloc((size_t)N * 4);
  float*          h      = (float*)alloc((size_t)N * H * 4);
  unsigned short* mb     = (unsigned short*)alloc((size_t)N * H * 2);    // bf16 GEMM A staging
  unsigned short* wt     = (unsigned short*)alloc((size_t)H * H * 2);    // bf16 B^T staging
  float*          xl     = (float*)alloc((size_t)N * H * 4);
  float*          agg    = (float*)alloc((size_t)N * H * 4);
  float*          Qb     = (float*)alloc((size_t)N * H * 4);
  float*          Kb     = (float*)alloc((size_t)N * H * 4);
  float*          Vb     = (float*)alloc((size_t)N * H * 4);
  float*          a_s    = (float*)alloc((size_t)N * NH * 4);
  float*          a_d    = (float*)alloc((size_t)N * NH * 4);
  float*          we     = (float*)alloc((size_t)BINS * NH * 4);
  float*          alpha  = (float*)alloc((size_t)EA * NH * 4);
  float*          segmax = (float*)alloc((size_t)N * NH * 4);
  float*          segden = (float*)alloc((size_t)N * NH * 4);
  float*          geo    = (float*)alloc((size_t)E * NH * 4);
  float*          pooled = (float*)alloc((size_t)G * H * 4);
  float*          gcnt   = (float*)alloc((size_t)G * 4);

  const int TB = 256;
  auto nb = [](long long n, int b) { return dim3((unsigned)((n + b - 1) / b)); };
  dim3 gemm_grid((unsigned)((N + 15) / 16), H / 32);   // 16x32 strip per wave

  // RBF features + self-loop mean edge features + node embedding
  k_fill<<<nb((long long)N * BINS, TB), TB, 0, stream>>>(efloop, 0.f, (long long)N * BINS);
  k_fill<<<nb(N, TB), TB, 0, stream>>>(cnt, 0.f, N);
  k_rbf<<<nb((long long)E * 64, TB), TB, 0, stream>>>(edge_attr, efb, E);
  k_loop_acc<<<nb((long long)E * BINS, TB), TB, 0, stream>>>(edge_attr, dst, efloop, cnt, E);
  k_loop_fin<<<nb((long long)N * 64, TB), TB, 0, stream>>>(efloop, cnt, efb, N, E);
  k_embed<<<nb((long long)N * H, TB), TB, 0, stream>>>(x, is_defect, atom_emb, defect_emb, h, N);

  // ---- GAT layers ----
  for (int l = 0; l < NL; l++) {
    k_tobf16<<<nb((long long)N * H, TB), TB, 0, stream>>>(h, mb, (long long)N * H);
    k_transpose_bf16<<<nb(H * H, TB), TB, 0, stream>>>(gat_w + (size_t)l * H * H, wt, H, H);
    k_gemm_bf16<<<gemm_grid, 32, 0, stream>>>(mb, wt, xl, nullptr, N, H, H);
    k_node_coef<<<nb((long long)N * NH, TB), TB, 0, stream>>>(xl, gat_as + l * NH * HD,
                                                              gat_ad + l * NH * HD, a_s, a_d, N);
    k_fold_we<<<1, 256, 0, stream>>>(gat_ew + (size_t)l * BINS * H, gat_ae + l * NH * HD, we);
    k_fill<<<nb((long long)N * NH, TB), TB, 0, stream>>>(segmax, -1e30f, (long long)N * NH);
    k_fill<<<nb((long long)N * NH, TB), TB, 0, stream>>>(segden, 0.f, (long long)N * NH);
    k_fill<<<nb((long long)N * H, TB), TB, 0, stream>>>(agg, 0.f, (long long)N * H);
    k_gat_score<<<nb((long long)EA * NH, TB), TB, 0, stream>>>(a_s, a_d, efb, we, src, dst,
                                                               alpha, segmax, E, EA);
    k_seg_exp<<<nb((long long)EA * NH, TB), TB, 0, stream>>>(alpha, dst, segmax, segden, E, EA);
    k_seg_norm<<<nb((long long)EA * NH, TB), TB, 0, stream>>>(alpha, dst, segden, E, EA);
    k_scatter<<<nb((long long)EA * H, TB), TB, 0, stream>>>(xl, alpha, src, dst, agg, E, EA);
    k_update_ln<<<N, H, 0, stream>>>(h, agg, gat_b + l * H, 1, N);
  }

  // ---- edge-transformer layers ----
  for (int l = 0; l < NL; l++) {
    k_tobf16<<<nb((long long)N * H, TB), TB, 0, stream>>>(h, mb, (long long)N * H);
    k_transpose_bf16<<<nb(H * H, TB), TB, 0, stream>>>(qw + (size_t)l * H * H, wt, H, H);
    k_gemm_bf16<<<gemm_grid, 32, 0, stream>>>(mb, wt, Qb, qb + l * H, N, H, H);
    k_transpose_bf16<<<nb(H * H, TB), TB, 0, stream>>>(kw + (size_t)l * H * H, wt, H, H);
    k_gemm_bf16<<<gemm_grid, 32, 0, stream>>>(mb, wt, Kb, kb + l * H, N, H, H);
    k_transpose_bf16<<<nb(H * H, TB), TB, 0, stream>>>(vw + (size_t)l * H * H, wt, H, H);
    k_gemm_bf16<<<gemm_grid, 32, 0, stream>>>(mb, wt, Vb, vb + l * H, N, H, H);
    k_geo<<<2048, 128, 0, stream>>>(efb, gw1 + (size_t)l * BINS * H, gb1 + l * H,
                                    gw2 + l * H * NH, gb2 + l * NH, geo, E);
    k_fill<<<nb((long long)N * NH, TB), TB, 0, stream>>>(segmax, -1e30f, (long long)N * NH);
    k_fill<<<nb((long long)N * NH, TB), TB, 0, stream>>>(segden, 0.f, (long long)N * NH);
    k_fill<<<nb((long long)N * H, TB), TB, 0, stream>>>(agg, 0.f, (long long)N * H);
    k_qk_score<<<nb((long long)E * NH, TB), TB, 0, stream>>>(Qb, Kb, geo, dbias + l * NH * 4,
                                                             is_defect, src, dst, alpha, segmax, E);
    k_seg_exp<<<nb((long long)E * NH, TB), TB, 0, stream>>>(alpha, dst, segmax, segden, E, E);
    k_seg_norm<<<nb((long long)E * NH, TB), TB, 0, stream>>>(alpha, dst, segden, E, E);
    k_scatter<<<nb((long long)E * H, TB), TB, 0, stream>>>(Vb, alpha, src, dst, agg, E, E);
    k_tobf16<<<nb((long long)N * H, TB), TB, 0, stream>>>(agg, mb, (long long)N * H);
    k_transpose_bf16<<<nb(H * H, TB), TB, 0, stream>>>(ow + (size_t)l * H * H, wt, H, H);
    k_gemm_bf16<<<gemm_grid, 32, 0, stream>>>(mb, wt, xl, ob + l * H, N, H, H);
    k_update_ln<<<N, H, 0, stream>>>(h, xl, nullptr, 0, N);
  }

  // ---- mean-pool per graph + MLP head ----
  k_fill<<<nb((long long)G * H, TB), TB, 0, stream>>>(pooled, 0.f, (long long)G * H);
  k_fill<<<nb(G, TB), TB, 0, stream>>>(gcnt, 0.f, G);
  k_pool<<<nb((long long)N * H, TB), TB, 0, stream>>>(h, batch, pooled, gcnt, N);
  k_mlp<<<G, H, 0, stream>>>(pooled, gcnt, fcw1, fcb1, fcw2, fcb2, (float*)d_out);
}
